// GRU_Model_32719060861682
// MI455X (gfx1250) — matrix operand order repaired and checked by reading, not compiled, not run on hardware
//
#include <hip/hip_runtime.h>
#include <hip/hip_bf16.h>

typedef _Float16 v16h __attribute__((ext_vector_type(16)));
typedef float    v8f  __attribute__((ext_vector_type(8)));

constexpr int IN   = 8;     // input dim
constexpr int HID  = 50;    // hidden dim
constexpr int G    = 150;   // 3*HID gate columns
constexpr int NP   = 160;   // gate columns padded to 10 WMMA N-tiles
constexpr int KX   = 32;    // K padding for input projection (IN=8 -> 32; row 8 carries b_ih)
constexpr int KH   = 64;    // K padding for hidden projection (HID=50 -> 64; row 50 carries b_hh)
constexpr int TT   = 512;   // timesteps
constexpr int BT   = 16;    // batch tile per block (WMMA M)
constexpr int NW   = 5;     // waves per block (2 N-tiles each -> 10 tiles)
constexpr int THREADS = NW * 32;

__device__ __forceinline__ float fast_sigmoid(float x) {
    // v_rcp_f32(1 + v_exp_f32(-x*log2e)) ; v_exp_f32 is exp2 on AMD
    return __builtin_amdgcn_rcpf(1.0f + __builtin_amdgcn_exp2f(-1.4426950408889634f * x));
}
__device__ __forceinline__ float fast_tanh(float x) {
    // tanh(x) = 1 - 2/(exp(2x)+1); saturates to +/-1 correctly at both infinities
    float e = __builtin_amdgcn_exp2f(2.8853901817779268f * x);
    return 1.0f - 2.0f * __builtin_amdgcn_rcpf(e + 1.0f);
}

__global__ __launch_bounds__(THREADS)
void gru_fused_kernel(const float* __restrict__ x,
                      const float* __restrict__ W_ih,
                      const float* __restrict__ W_hh,
                      const float* __restrict__ b_ih,
                      const float* __restrict__ b_hh,
                      const float* __restrict__ W_out,
                      const float* __restrict__ b_out,
                      float* __restrict__ out)
{
    // ---- LDS residency (≈53 KB) ----
    __shared__ _Float16 sWih[KX][NP];   // W_ih^T f16, zero padded; row k=8  = b_ih (rank-1 bias fold)
    __shared__ _Float16 sWhh[KH][NP];   // W_hh^T f16, zero padded; row k=50 = b_hh (rank-1 bias fold)
    __shared__ _Float16 shh[BT][KH];    // h in f16, A-frag row layout; col 50 pinned to 1.0 (bias lane)
    __shared__ float    sgi[BT][NP];    // x_t @ W_ih^T + b_ih    (also reused as head scratch at end)
    __shared__ float    sgh[BT][NP];    // h   @ W_hh^T + b_hh

    const int tid  = threadIdx.x;
    const int lane = tid & 31;
    const int wid  = tid >> 5;
    const int m16  = lane & 15;   // M (A/C rows) / N (B/C cols) index
    const int half = lane >> 4;   // lane-half selector in WMMA layouts
    const int b0   = blockIdx.x * BT;

    // ---- one-time setup: repack weights (transposed, f16, padded, biases folded in) ----
    for (int i = tid; i < KX * NP; i += THREADS) {
        int k = i / NP, n = i % NP;
        float v = 0.0f;
        if (n < G) {
            if (k < IN)       v = W_ih[n * IN + k];
            else if (k == IN) v = b_ih[n];          // A[m][8] = 1 adds b_ih via WMMA
        }
        sWih[k][n] = (_Float16)v;
    }
    for (int i = tid; i < KH * NP; i += THREADS) {
        int k = i / NP, n = i % NP;
        float v = 0.0f;
        if (n < G) {
            if (k < HID)       v = W_hh[n * HID + k];
            else if (k == HID) v = b_hh[n];         // A[m][50] = 1 adds b_hh via WMMA
        }
        sWhh[k][n] = (_Float16)v;
    }
    for (int i = tid; i < BT * KH; i += THREADS) {
        // h = 0; the K=HID padding column is pinned to 1.0 and never rewritten
        (&shh[0][0])[i] = (_Float16)((i % KH) == HID ? 1.0f : 0.0f);
    }
    __syncthreads();

    // ---- loop-invariant B fragments in VGPRs (ISA B 32x16 f16: k = (e<8 ? half*8 + e : 16 + half*8 + e-8), n = lane&15) ----
    const int t0 = wid * 2, t1 = wid * 2 + 1;
    const int n0 = t0 * 16 + m16;
    const int n1 = t1 * 16 + m16;
    v16h bi0, bi1, bh00, bh01, bh10, bh11;
    #pragma unroll
    for (int e = 0; e < 16; ++e) {
        const int kx = (e < 8) ? (half * 8 + e) : (16 + half * 8 + (e - 8));
        bi0[e]  = sWih[kx][n0];
        bi1[e]  = sWih[kx][n1];
        bh00[e] = sWhh[kx][n0];
        bh01[e] = sWhh[kx][n1];
        bh10[e] = sWhh[32 + kx][n0];
        bh11[e] = sWhh[32 + kx][n1];
    }

    // ---- per-thread private hidden state (f32, full precision on the z*h recurrence) ----
    float hreg[5];
    #pragma unroll
    for (int i = 0; i < 5; ++i) hreg[i] = 0.0f;

    const float* xrow = x + (size_t)(b0 + m16) * TT * IN;

    for (int t = 0; t < TT; ++t) {
        // ---- A fragment for x_t: lanes 0-15 carry K=0..7 (inputs); lanes 16-31 K=8 carries 1.0 ----
        v16h ax = {};
        if (half == 0) {
            const float* xr = xrow + (size_t)t * IN;
            #pragma unroll
            for (int e = 0; e < 8; ++e) ax[e] = (_Float16)xr[e];
            if (t + 8 < TT) __builtin_prefetch(xr + 8 * IN, 0, 3);  // near-cache prefetch down the stream
        } else {
            ax[0] = (_Float16)1.0f;   // K=8 bias lane
        }

        // ---- A fragments for h: two 16x32 tiles (K 0..31, 32..63), contiguous 8-half runs ----
        v16h ah0, ah1;
        {
            const _Float16* hr = &shh[m16][0];
            const int o = half * 8;
            #pragma unroll
            for (int e = 0; e < 8; ++e) {
                ah0[e]     = hr[o + e];
                ah0[8 + e] = hr[16 + o + e];
                ah1[e]     = hr[32 + o + e];
                ah1[8 + e] = hr[48 + o + e];
            }
        }

        // ---- 6 WMMAs per wave per step (f16 in, f32 accumulate; biases ride along) ----
        v8f ci0 = {}, ci1 = {}, ch0 = {}, ch1 = {};
        ci0 = __builtin_amdgcn_wmma_f32_16x16x32_f16(false, ax,  false, bi0,  (short)0, ci0, false, false);
        ci1 = __builtin_amdgcn_wmma_f32_16x16x32_f16(false, ax,  false, bi1,  (short)0, ci1, false, false);
        ch0 = __builtin_amdgcn_wmma_f32_16x16x32_f16(false, ah0, false, bh00, (short)0, ch0, false, false);
        ch1 = __builtin_amdgcn_wmma_f32_16x16x32_f16(false, ah0, false, bh01, (short)0, ch1, false, false);
        ch0 = __builtin_amdgcn_wmma_f32_16x16x32_f16(false, ah1, false, bh10, (short)0, ch0, false, false);
        ch1 = __builtin_amdgcn_wmma_f32_16x16x32_f16(false, ah1, false, bh11, (short)0, ch1, false, false);

        // ---- spill C tiles to LDS (C layout: m = v + 8*half, n = lane&15) ----
        #pragma unroll
        for (int v = 0; v < 8; ++v) {
            const int mm = v + 8 * half;
            sgi[mm][n0] = ci0[v];
            sgi[mm][n1] = ci1[v];
            sgh[mm][n0] = ch0[v];
            sgh[mm][n1] = ch1[v];
        }
        __syncthreads();

        // ---- GRU gate math: 16*50 = 800 elements over 160 threads; h lives in registers ----
        #pragma unroll
        for (int i = 0; i < 5; ++i) {
            const int idx = tid + i * THREADS;        // 0..799 (fixed mapping for all t)
            const int mm = idx / HID, j = idx % HID;
            const float gr  = sgi[mm][j]           + sgh[mm][j];
            const float gz  = sgi[mm][HID + j]     + sgh[mm][HID + j];
            const float gin = sgi[mm][2 * HID + j];
            const float ghn = sgh[mm][2 * HID + j];
            const float r  = fast_sigmoid(gr);
            const float z  = fast_sigmoid(gz);
            const float nn = fast_tanh(gin + r * ghn);
            hreg[i] = (1.0f - z) * nn + z * hreg[i];
            shh[mm][j] = (_Float16)hreg[i];
        }
        __syncthreads();
    }

    // ---- spill final h (f32) into sgi scratch, then linear head ----
    #pragma unroll
    for (int i = 0; i < 5; ++i) {
        const int idx = tid + i * THREADS;
        sgi[idx / HID][idx % HID] = hreg[i];
    }
    __syncthreads();
    if (tid < BT) {
        float acc = b_out[0];
        #pragma unroll
        for (int j = 0; j < HID; ++j) acc += sgi[tid][j] * W_out[j];
        out[b0 + tid] = acc;
    }
}

extern "C" void kernel_launch(void* const* d_in, const int* in_sizes, int n_in,
                              void* d_out, int out_size, void* d_ws, size_t ws_size,
                              hipStream_t stream) {
    const float* x     = (const float*)d_in[0];
    const float* W_ih  = (const float*)d_in[1];
    const float* W_hh  = (const float*)d_in[2];
    const float* b_ih  = (const float*)d_in[3];
    const float* b_hh  = (const float*)d_in[4];
    const float* W_out = (const float*)d_in[5];
    const float* b_out = (const float*)d_in[6];
    float* out = (float*)d_out;

    const int B = in_sizes[0] / (TT * IN);   // 4096
    dim3 grid(B / BT), block(THREADS);       // 256 blocks x 160 threads (5 waves)
    gru_fused_kernel<<<grid, block, 0, stream>>>(x, W_ih, W_hh, b_ih, b_hh, W_out, b_out, out);
}
